// MultiLayerModularity_25555055411452
// MI455X (gfx1250) — matrix operand order repaired and checked by reading, not compiled, not run on hardware
//
#include <hip/hip_runtime.h>
#include <hip/hip_bf16.h>
#include <math.h>

typedef __attribute__((ext_vector_type(16))) _Float16 v16h;
typedef __attribute__((ext_vector_type(8)))  float    v8f;
typedef unsigned int v4u __attribute__((ext_vector_type(4)));
typedef int          v8i __attribute__((ext_vector_type(8)));
typedef int          v4i __attribute__((ext_vector_type(4)));

#define THREADS 256
#define MAX_KPAD (9 * 128)   // conv4: Cin=128 -> Kpad=1152

#if defined(__has_builtin)
#if __has_builtin(__builtin_amdgcn_tensor_load_to_lds)
#define HAVE_TDM 1
#endif
#endif

// ------------------------------------------------------------------ utilities
__global__ void zero_f32(float* __restrict__ p, int n) {
  int i = blockIdx.x * blockDim.x + threadIdx.x;
  if (i < n) p[i] = 0.f;
}

__global__ void zero_f16(_Float16* __restrict__ p, size_t n) {
  for (size_t i = (size_t)blockIdx.x * blockDim.x + threadIdx.x; i < n;
       i += (size_t)gridDim.x * blockDim.x)
    p[i] = (_Float16)0.f;
}

// images (N,3,H,W) f32 -> zero-haloed NHWC f16 (N,H+2,W+2,Cp) with Cp=32 padded channels
__global__ void img_to_nhwc_pad(const float* __restrict__ img, _Float16* __restrict__ out,
                                int N, int H, int W, int Cp) {
  size_t tot = (size_t)N * (H + 2) * (W + 2) * Cp;
  for (size_t i = (size_t)blockIdx.x * blockDim.x + threadIdx.x; i < tot;
       i += (size_t)gridDim.x * blockDim.x) {
    int c = (int)(i % Cp); size_t t = i / Cp;
    int w = (int)(t % (W + 2)); t /= (W + 2);
    int h = (int)(t % (H + 2)); int n = (int)(t / (H + 2));
    float v = 0.f;
    if (c < 3 && h >= 1 && h <= H && w >= 1 && w <= W)
      v = img[(((size_t)n * 3 + c) * H + (h - 1)) * W + (w - 1)];
    out[i] = (_Float16)v;
  }
}

// pack OIHW f32 weights into (Cout, 9*CinPad) f16 rows, tap-major channel-minor:
// wp[co][rs*CinPad + ci] so each 32-wide WMMA K-block lies inside one (r,s) tap.
__global__ void pack_weights(const float* __restrict__ w, _Float16* __restrict__ wp,
                             int Cout, int Cin, int CinPad) {
  int Kpad = 9 * CinPad;
  int tot = Cout * Kpad;
  for (int i = blockIdx.x * blockDim.x + threadIdx.x; i < tot; i += gridDim.x * blockDim.x) {
    int co = i / Kpad, k = i - co * Kpad;
    int rs = k / CinPad, ci = k - rs * CinPad;
    wp[i] = (ci < Cin) ? (_Float16)w[(co * Cin + ci) * 9 + rs] : (_Float16)0.f;
  }
}

// ------------------------------------------------- implicit-GEMM 3x3 conv (WMMA)
// Activations NHWC f16 with zero 1-pixel halo: X is (N,H+2,W+2,Cin), interior at (1,1).
// Block = 32 cout x 256 pixels (8 waves x 32 pixels). Weight tile (32 x Kpad f16) is
// staged to LDS by the Tensor Data Mover once per block. Inner loop per 32-wide
// K-block: 4x global b128 (2 B frags), 4x ds b128 (2 A frags), 4x v_wmma (2x2 tile).
// No masking anywhere: halo makes every tap address valid and correct.
__global__ void __launch_bounds__(THREADS)
conv3x3_wmma(const _Float16* __restrict__ X,   // (N,H+2,W+2,Cin) f16, zero halo
             const _Float16* __restrict__ Wg,  // (Cout, 9*Cin) f16 packed
             const float* __restrict__ bias,   // (Cout)
             _Float16* __restrict__ Y,         // NHWC f16 (haloed if outPad)
             int N, int Cin, int Cout, int H, int W, int outPad) {
  // NOTE: smw must be the kernel's only LDS allocation (TDM descriptor uses lds_addr=0)
  __shared__ _Float16 smw[32 * MAX_KPAD];

  const int Kpad  = 9 * Cin;
  const int lane  = threadIdx.x & 31;
  const int wv    = threadIdx.x >> 5;
  const int mblks = Cout >> 5;
  const int HW    = H * W;
  const int Wp    = W + 2;

  const int mblk = blockIdx.x % mblks;
  const int pblk = blockIdx.x / mblks;
  const int m0   = mblk << 5;                 // 32 output channels per block
  const int p0   = (pblk << 8) + (wv << 5);   // 32 pixels per wave (W % 32 == 0)
  const int n    = p0 / HW;
  const int hw   = p0 - n * HW;
  const int h    = hw / W;
  const int w0   = hw - h * W;

  const int hi   = lane >> 4;
  const int col  = lane & 15;
  const int wpix = w0 + col;                  // pixel tile 0; tile 1 at +16

  // ---- stage 32 x Kpad weight tile into LDS ----
  const _Float16* wtile = Wg + (size_t)m0 * Kpad;
#if HAVE_TDM
  if (threadIdx.x < 32) {  // wave 0 issues the DMA (TDM ignores EXEC; one issue/wave)
    unsigned long long ga = (unsigned long long)(const void*)wtile;
    unsigned T = 32u * (unsigned)Kpad;        // tile elements (f16)
    // D# group0: count=1 (user), lds_addr=0, 57b global byte addr, type=2
    v4u g0 = { 1u, 0u, (unsigned)(ga & 0xFFFFFFFFu),
               (unsigned)(((ga >> 32) & 0x01FFFFFFu) | (2u << 30)) };
    // D# group1: mask=0, data_size=1(2B); 1-D tile: tensor_dim0=tile_dim0=T
    v8i g1 = { (int)(1u << 16),
               (int)((T & 0xFFFFu) << 16),                 // tensor_dim0[15:0]
               (int)(((T >> 16) & 0xFFFFu) | (1u << 16)),  // tensor_dim0[31:16] | tensor_dim1=1
               (int)((T & 0xFFFFu) << 16),                 // tile_dim0
               0,                                          // tile_dim1/2 unused
               (int)T, 0, 0 };                             // tensor_dim0_stride
    v4i z4 = { 0, 0, 0, 0 };
#if __clang_major__ >= 23
    v8i z8 = { 0, 0, 0, 0, 0, 0, 0, 0 };
    __builtin_amdgcn_tensor_load_to_lds(g0, g1, z4, z4, z8, 0);
#else
    __builtin_amdgcn_tensor_load_to_lds(g0, g1, z4, z4, 0);
#endif
    __builtin_amdgcn_s_wait_tensorcnt(0);
  }
  // Escape smw into an asm with a memory clobber: the TDM write to LDS is opaque to
  // the compiler; without this it folds all smw loads to undef and deletes the A path.
  asm volatile("" : : "r"((void*)smw) : "memory");
#else
  {  // cooperative staging fallback
    const unsigned long long* src = (const unsigned long long*)wtile;
    unsigned long long* dst = (unsigned long long*)smw;
    int cnt = (32 * Kpad) >> 2;
    for (int i = threadIdx.x; i < cnt; i += THREADS) dst[i] = src[i];
  }
#endif
  __syncthreads();

  union V16 { v16h v; uint4 q[2]; };
  v8f acc00 = {}, acc01 = {}, acc10 = {}, acc11 = {};
  const _Float16* xn = X + (size_t)n * (H + 2) * Wp * Cin;

  for (int rs = 0; rs < 9; ++rs) {
    const int r = rs / 3, s = rs - (rs / 3) * 3;
    // padded coords (h+r, wpix+s) are always in-bounds; halo holds zeros
    const _Float16* xp0 = xn + ((size_t)(h + r) * Wp + (wpix + s)) * Cin + hi * 16;
    const _Float16* xp1 = xp0 + 16 * Cin;     // pixel tile 1 (wpix+16)
    const _Float16* wr0 = &smw[(size_t)col * Kpad + rs * Cin];
    const _Float16* wr1 = &smw[(size_t)(col + 16) * Kpad + rs * Cin];
    __builtin_prefetch(xp0 + (size_t)Wp * Cin, 0, 3);  // global_prefetch_b8 (next tap row)

    for (int cb = 0; cb < Cin; cb += 32) {
      // B fragments: 16 contiguous channels at each pixel tile -> 2x b128 each
      V16 B0, B1;
      B0.q[0] = *(const uint4*)(xp0 + cb);
      B0.q[1] = *(const uint4*)(xp0 + cb + 8);
      B1.q[0] = *(const uint4*)(xp1 + cb);
      B1.q[1] = *(const uint4*)(xp1 + cb + 8);
      // A fragments from LDS (contiguous layout: lane<16 K {0..15}, lane>=16 {16..31})
      V16 A0, A1;
      A0.q[0] = *(const uint4*)(wr0 + cb + hi * 16);
      A0.q[1] = *(const uint4*)(wr0 + cb + hi * 16 + 8);
      A1.q[0] = *(const uint4*)(wr1 + cb + hi * 16);
      A1.q[1] = *(const uint4*)(wr1 + cb + hi * 16 + 8);
      acc00 = __builtin_amdgcn_wmma_f32_16x16x32_f16(false, A0.v, false, B0.v, (short)0,
                                                     acc00, false, false);
      acc01 = __builtin_amdgcn_wmma_f32_16x16x32_f16(false, A1.v, false, B0.v, (short)0,
                                                     acc01, false, false);
      acc10 = __builtin_amdgcn_wmma_f32_16x16x32_f16(false, A0.v, false, B1.v, (short)0,
                                                     acc10, false, false);
      acc11 = __builtin_amdgcn_wmma_f32_16x16x32_f16(false, A1.v, false, B1.v, (short)0,
                                                     acc11, false, false);
    }
  }

  // store: lane's 8 rows per tile are contiguous channels -> one b128 per tile
  const int    Wrow    = outPad ? Wp : W;
  const size_t nStrOut = (size_t)(outPad ? (H + 2) * Wp : HW);
  const size_t pix0 = (size_t)n * nStrOut +
                      (outPad ? ((size_t)(h + 1) * Wrow + (wpix + 1))
                              : ((size_t)h * Wrow + wpix));
  _Float16* yp0 = Y + pix0 * Cout + m0 + hi * 8;
  _Float16* yp1 = yp0 + (size_t)16 * Cout;    // pixel tile 1
  float b0[8], b1[8];
#pragma unroll
  for (int j = 0; j < 8; ++j) {
    b0[j] = bias[m0 + hi * 8 + j];
    b1[j] = bias[m0 + 16 + hi * 8 + j];
  }
  _Float16 o[8];
#pragma unroll
  for (int j = 0; j < 8; ++j) { float y = acc00[j] + b0[j]; o[j] = (_Float16)(y > 0.f ? y : 0.f); }
  *(uint4*)yp0 = *(const uint4*)o;
#pragma unroll
  for (int j = 0; j < 8; ++j) { float y = acc01[j] + b1[j]; o[j] = (_Float16)(y > 0.f ? y : 0.f); }
  *(uint4*)(yp0 + 16) = *(const uint4*)o;
#pragma unroll
  for (int j = 0; j < 8; ++j) { float y = acc10[j] + b0[j]; o[j] = (_Float16)(y > 0.f ? y : 0.f); }
  *(uint4*)yp1 = *(const uint4*)o;
#pragma unroll
  for (int j = 0; j < 8; ++j) { float y = acc11[j] + b1[j]; o[j] = (_Float16)(y > 0.f ? y : 0.f); }
  *(uint4*)(yp1 + 16) = *(const uint4*)o;
}

// ------------------------------------------------------------------ pooling
// NHWC, haloed in and out: x (N,H+2,W+2,C) -> y (N,H/2+2,W/2+2,C), interiors at (1,1)
__global__ void maxpool2_nhwc(const _Float16* __restrict__ x, _Float16* __restrict__ y,
                              int N, int C, int H, int W) {
  int Ho = H >> 1, Wo = W >> 1;
  int Wp = W + 2, Wop = Wo + 2;
  size_t tot = (size_t)N * Ho * Wo * C;
  for (size_t i = (size_t)blockIdx.x * blockDim.x + threadIdx.x; i < tot;
       i += (size_t)gridDim.x * blockDim.x) {
    int c = (int)(i % C); size_t t = i / C;
    int wo = (int)(t % Wo); t /= Wo;
    int ho = (int)(t % Ho); int n = (int)(t / Ho);
    const _Float16* p = x + (((size_t)n * (H + 2) + 2 * ho + 1) * Wp + 2 * wo + 1) * C + c;
    size_t rstep = (size_t)Wp * C;
    float a = (float)p[0], b = (float)p[C], cc = (float)p[rstep], d = (float)p[rstep + C];
    y[(((size_t)n * (Ho + 2) + ho + 1) * Wop + wo + 1) * C + c] =
        (_Float16)fmaxf(fmaxf(a, b), fmaxf(cc, d));
  }
}

__global__ void avgpool2_f32(const float* __restrict__ x, float* __restrict__ y,
                             int N, int C, int H, int W) {
  int Ho = H >> 1, Wo = W >> 1;
  size_t tot = (size_t)N * C * Ho * Wo;
  for (size_t i = (size_t)blockIdx.x * blockDim.x + threadIdx.x; i < tot;
       i += (size_t)gridDim.x * blockDim.x) {
    int wo = (int)(i % Wo); size_t t = i / Wo;
    int ho = (int)(t % Ho); t /= Ho;
    int c  = (int)(t % C);  int n = (int)(t / C);
    const float* p = x + (((size_t)n * C + c) * H + 2 * ho) * W + 2 * wo;
    y[i] = 0.25f * (p[0] + p[1] + p[W] + p[W + 1]);
  }
}

// ------------------------------------------------------------------ batchnorm
// NHWC f16 stats: one block per channel, contiguous-c layout -> divide-free loop
__global__ void bn_stats_nhwc(const _Float16* __restrict__ x, float* __restrict__ mi,
                              int NHW, int C) {
  int c = blockIdx.x;
  float s = 0.f, s2 = 0.f;
  for (int i = threadIdx.x; i < NHW; i += blockDim.x) {
    float v = (float)x[(size_t)i * C + c];
    s += v; s2 += v * v;
  }
  __shared__ float bs[THREADS], bs2[THREADS];
  bs[threadIdx.x] = s; bs2[threadIdx.x] = s2;
  __syncthreads();
  for (int off = blockDim.x >> 1; off > 0; off >>= 1) {
    if ((int)threadIdx.x < off) {
      bs[threadIdx.x]  += bs[threadIdx.x + off];
      bs2[threadIdx.x] += bs2[threadIdx.x + off];
    }
    __syncthreads();
  }
  if (threadIdx.x == 0) {
    float m   = bs[0] / (float)NHW;
    float var = bs2[0] / (float)NHW - m * m;   // biased variance (jnp.var ddof=0)
    mi[2 * c]     = m;
    mi[2 * c + 1] = rsqrtf(var + 1e-5f);
  }
}

// normalize NHWC f16 -> NCHW f32 (modularity wants lane-coalesced NCHW reads)
__global__ void bn_apply_nhwc2nchw(const _Float16* __restrict__ x, const float* __restrict__ mi,
                                   float* __restrict__ y, int N, int C, int HW) {
  size_t tot = (size_t)N * C * HW;
  for (size_t i = (size_t)blockIdx.x * blockDim.x + threadIdx.x; i < tot;
       i += (size_t)gridDim.x * blockDim.x) {
    int p = (int)(i % HW); size_t t = i / HW;
    int c = (int)(t % C);  int n = (int)(t / C);
    y[i] = ((float)x[((size_t)n * HW + p) * C + c] - mi[2 * c]) * mi[2 * c + 1];
  }
}

// f32 NCHW stats (images): n-outer loop, no divides
__global__ void bn_stats_nchw(const float* __restrict__ x, float* __restrict__ mi,
                              int N, int C, int HW) {
  int c = blockIdx.x;
  float s = 0.f, s2 = 0.f;
  for (int nn = 0; nn < N; ++nn) {
    const float* p = x + ((size_t)nn * C + c) * HW;
    for (int i = threadIdx.x; i < HW; i += blockDim.x) {
      float v = p[i]; s += v; s2 += v * v;
    }
  }
  __shared__ float bs[THREADS], bs2[THREADS];
  bs[threadIdx.x] = s; bs2[threadIdx.x] = s2;
  __syncthreads();
  for (int off = blockDim.x >> 1; off > 0; off >>= 1) {
    if ((int)threadIdx.x < off) {
      bs[threadIdx.x]  += bs[threadIdx.x + off];
      bs2[threadIdx.x] += bs2[threadIdx.x + off];
    }
    __syncthreads();
  }
  if (threadIdx.x == 0) {
    float tot = (float)(N * HW);
    float m   = bs[0] / tot;
    float var = bs2[0] / tot - m * m;
    mi[2 * c]     = m;
    mi[2 * c + 1] = rsqrtf(var + 1e-5f);
  }
}

__global__ void bn_apply_nchw(const float* __restrict__ x, const float* __restrict__ mi,
                              float* __restrict__ y, int N, int C, int HW) {
  size_t tot = (size_t)N * C * HW;
  for (size_t i = (size_t)blockIdx.x * blockDim.x + threadIdx.x; i < tot;
       i += (size_t)gridDim.x * blockDim.x) {
    int c = (int)((i / (size_t)HW) % (size_t)C);
    y[i] = (x[i] - mi[2 * c]) * mi[2 * c + 1];
  }
}

// ------------------------------------------------------------------ modularity
__global__ void modularity_k(const float* __restrict__ x,    // (N,C,H,W) f32
                             const float* __restrict__ seg,  // (N,K,H,W) f32
                             float* __restrict__ num, float* __restrict__ den,
                             int N, int C, int K, int H, int W, float inv2s2) {
  const int HW = H * W;
  const int blocksPerN = HW / THREADS;
  const int n    = blockIdx.x / blocksPerN;
  const int pix  = (blockIdx.x % blocksPerN) * THREADS + threadIdx.x;
  const int h    = pix / W, w = pix - h * W;

  __shared__ float snum[16], sden[16];
  if ((int)threadIdx.x < K) { snum[threadIdx.x] = 0.f; sden[threadIdx.x] = 0.f; }
  __syncthreads();

  int off[25];
#pragma unroll
  for (int o = 0; o < 25; ++o) {
    int di = o / 5 - 2, dj = o % 5 - 2;
    int h2 = h + di, w2 = w + dj;
    off[o] = (h2 >= 0 && h2 < H && w2 >= 0 && w2 < W) ? (h2 * W + w2) : -1;
  }

  float arr[25];
#pragma unroll
  for (int o = 0; o < 25; ++o) arr[o] = 0.f;

  const float* xb = x + (size_t)n * C * HW;
  for (int c = 0; c < C; ++c) {
    const float* xc = xb + (size_t)c * HW;
    float xv = xc[pix];
#pragma unroll
    for (int o = 0; o < 25; ++o) {
      if (off[o] >= 0) { float d = xv - xc[off[o]]; arr[o] += d * d; }
    }
  }

  float w1 = 0.f;
#pragma unroll
  for (int o = 0; o < 25; ++o) {
    float wvv = (off[o] >= 0) ? __expf(-arr[o] * inv2s2) : 0.f;
    arr[o] = wvv;
    w1 += wvv;
  }

  const float* sb = seg + (size_t)n * K * HW;
  for (int k = 0; k < K; ++k) {
    const float* sk = sb + (size_t)k * HW;
    float s = sk[pix];
    float a = 0.f;
#pragma unroll
    for (int o = 0; o < 25; ++o)
      if (off[o] >= 0) a += sk[off[o]] * arr[o];
    atomicAdd(&snum[k], s * a);   // ds_add_f32
    atomicAdd(&sden[k], s * w1);
  }
  __syncthreads();
  if ((int)threadIdx.x < K) {
    atomicAdd(&num[n * K + threadIdx.x], snum[threadIdx.x]);
    atomicAdd(&den[n * K + threadIdx.x], sden[threadIdx.x]);
  }
}

__global__ void finalize_k(const float* __restrict__ acc, float* __restrict__ out,
                           int N, int K) {
  if (threadIdx.x == 0 && blockIdx.x == 0) {
    const float* numV = acc;      const float* denV = acc + 32;
    const float* numI = acc + 64; const float* denI = acc + 96;
    float lv = 0.f, li = 0.f;
    for (int i = 0; i < N * K; ++i) lv += numV[i] / denV[i];
    for (int i = 0; i < N * K; ++i) li += numI[i] / denI[i];
    float inv = 1.0f / (float)(K * N);
    out[0] = (lv * inv + li * inv) * 0.5f;   // / NORM
  }
}

// ------------------------------------------------------------------ launcher
extern "C" void kernel_launch(void* const* d_in, const int* in_sizes, int n_in,
                              void* d_out, int out_size, void* d_ws, size_t ws_size,
                              hipStream_t stream) {
  (void)in_sizes; (void)n_in; (void)out_size; (void)ws_size;
  const float* images = (const float*)d_in[0];   // (4,3,128,128)
  const float* seg    = (const float*)d_in[1];   // (4,8,128,128)
  const float* w1 = (const float*)d_in[2]; const float* b1 = (const float*)d_in[3];
  const float* w2 = (const float*)d_in[4]; const float* b2 = (const float*)d_in[5];
  const float* w3 = (const float*)d_in[6]; const float* b3 = (const float*)d_in[7];
  const float* w4 = (const float*)d_in[8]; const float* b4 = (const float*)d_in[9];
  float* out = (float*)d_out;

  // ---- workspace carve-up (256B aligned) ----
  char* ws = (char*)d_ws;
  size_t cur = 0;
  auto alloc = [&](size_t bytes) -> char* {
    char* p = ws + cur;
    cur += (bytes + 255) & ~(size_t)255;
    return p;
  };
  const int Nb = 4;
  const size_t P130 = 130, P66 = 66;
  _Float16* x0p = (_Float16*)alloc((size_t)Nb * P130 * P130 * 32 * 2);   // images NHWC pad
  _Float16* w1p = (_Float16*)alloc((size_t)64 * 9 * 32 * 2);
  _Float16* w2p = (_Float16*)alloc((size_t)64 * 9 * 64 * 2);
  _Float16* w3p = (_Float16*)alloc((size_t)128 * 9 * 64 * 2);
  _Float16* w4p = (_Float16*)alloc((size_t)128 * 9 * 128 * 2);
  _Float16* y1p = (_Float16*)alloc((size_t)Nb * P130 * P130 * 64 * 2);
  _Float16* y2f = (_Float16*)alloc((size_t)Nb * P130 * P130 * 64 * 2);
  _Float16* y2s = (_Float16*)alloc((size_t)Nb * P66 * P66 * 64 * 2);
  _Float16* y3p = (_Float16*)alloc((size_t)Nb * P66 * P66 * 128 * 2);
  _Float16* y4  = (_Float16*)alloc((size_t)Nb * 64 * 64 * 128 * 2);     // unpadded NHWC
  float*    fich  = (float*)alloc((size_t)Nb * 128 * 64 * 64 * 4);      // NCHW
  float*    imgn  = (float*)alloc((size_t)Nb * 3 * 128 * 128 * 4);
  float*    segd  = (float*)alloc((size_t)Nb * 8 * 64 * 64 * 4);
  float*    stats = (float*)alloc(512 * 4);
  float*    acc   = (float*)alloc(128 * 4);

  // ---- layout conversion, halo zeroing, weight packing ----
  img_to_nhwc_pad<<<2048, THREADS, 0, stream>>>(images, x0p, Nb, 128, 128, 32);
  zero_f16<<<2048, THREADS, 0, stream>>>(y1p, (size_t)Nb * P130 * P130 * 64);
  zero_f16<<<2048, THREADS, 0, stream>>>(y2f, (size_t)Nb * P130 * P130 * 64);
  zero_f16<<<1024, THREADS, 0, stream>>>(y2s, (size_t)Nb * P66 * P66 * 64);
  zero_f16<<<1024, THREADS, 0, stream>>>(y3p, (size_t)Nb * P66 * P66 * 128);
  pack_weights<<<72,  THREADS, 0, stream>>>(w1, w1p, 64, 3, 32);
  pack_weights<<<144, THREADS, 0, stream>>>(w2, w2p, 64, 64, 64);
  pack_weights<<<288, THREADS, 0, stream>>>(w3, w3p, 128, 64, 64);
  pack_weights<<<576, THREADS, 0, stream>>>(w4, w4p, 128, 128, 128);

  // ---- VGG8 forward (WMMA convs, relu fused); blocks = (Cout/32)*(N*H*W/256) ----
  conv3x3_wmma<<<512, THREADS, 0, stream>>>(x0p, w1p, b1, y1p, Nb, 32, 64, 128, 128, 1);
  conv3x3_wmma<<<512, THREADS, 0, stream>>>(y1p, w2p, b2, y2f, Nb, 64, 64, 128, 128, 1);
  maxpool2_nhwc<<<2048, THREADS, 0, stream>>>(y2f, y2s, Nb, 64, 128, 128);
  conv3x3_wmma<<<256, THREADS, 0, stream>>>(y2s, w3p, b3, y3p, Nb, 64, 128, 64, 64, 1);
  conv3x3_wmma<<<256, THREADS, 0, stream>>>(y3p, w4p, b4, y4, Nb, 128, 128, 64, 64, 0);

  // ---- batchnorms ----
  bn_stats_nhwc<<<128, THREADS, 0, stream>>>(y4, stats, Nb * 64 * 64, 128);
  bn_apply_nhwc2nchw<<<2048, THREADS, 0, stream>>>(y4, stats, fich, Nb, 128, 64 * 64);
  bn_stats_nchw<<<3, THREADS, 0, stream>>>(images, stats + 256, Nb, 3, 128 * 128);
  bn_apply_nchw<<<768, THREADS, 0, stream>>>(images, stats + 256, imgn, Nb, 3, 128 * 128);

  // ---- seg avgpool 128 -> 64 ----
  avgpool2_f32<<<512, THREADS, 0, stream>>>(seg, segd, Nb, 8, 128, 128);

  // ---- modularity terms ----
  zero_f32<<<1, 128, 0, stream>>>(acc, 128);
  // sigma_vgg = 0.02 -> 1/(2*s^2) = 1250 ; sigma_img = 0.2 -> 12.5
  modularity_k<<<Nb * (64 * 64 / THREADS), THREADS, 0, stream>>>(
      fich, segd, acc, acc + 32, Nb, 128, 8, 64, 64, 1250.0f);
  modularity_k<<<Nb * (128 * 128 / THREADS), THREADS, 0, stream>>>(
      imgn, seg, acc + 64, acc + 96, Nb, 3, 8, 128, 128, 12.5f);

  finalize_k<<<1, 32, 0, stream>>>(acc, out, Nb, 8);
}
